// Conv2dWeightModulate_14886356648550
// MI455X (gfx1250) — compile-verified, matrix operand order checked by reading, not run on hardware
//
#include <hip/hip_runtime.h>
#include <stdint.h>

// ---------------------------------------------------------------------------
// StyleGAN2 modulated conv (B=8, 512->512, 64x64, 3x3) for gfx1250 (MI455X).
// Conv = implicit GEMM per batch: M=512, N=4096 (HW), K=4608 (cin*9),
// V_WMMA_F32_16X16X32_F16, A-tiles staged with global_load_async_to_lds_b128
// (double-buffered, fenced with s_wait_asynccnt).
// ---------------------------------------------------------------------------

typedef __attribute__((ext_vector_type(16))) _Float16 v16h;
typedef __attribute__((ext_vector_type(8)))  _Float16 v8h;
typedef __attribute__((ext_vector_type(8)))  float    v8f;

#define BATCH 8
#define CIN   512
#define COUT  512
#define HH    64
#define WW    64
#define KS    3
#define WDIM  512
#define KTOT  (CIN * KS * KS)   // 4608
#define NTOT  (HH * WW)         // 4096

#define C_LIN   0.04419417382415922f    // 1/sqrt(512)
#define C_CONV  0.014731391274719739f   // 1/sqrt(4608)
#define C_CONV2 (1.0f / 4608.0f)
#define EPS     1e-8f

// ---------------------------------------------------------------------------
// 1) style: s[b,c] = (sum_d w[b,d] * lin_w[c,d]) * c_lin + bias[c]
// ---------------------------------------------------------------------------
__global__ void style_kernel(const float* __restrict__ w,
                             const float* __restrict__ lw,
                             const float* __restrict__ lb,
                             float* __restrict__ s) {
    int idx = blockIdx.x * blockDim.x + threadIdx.x;   // 0..4095
    int b = idx / CIN;
    int c = idx % CIN;
    const float* wr = w  + (size_t)b * WDIM;
    const float* lr = lw + (size_t)c * WDIM;
    float acc = 0.0f;
    for (int d = 0; d < WDIM; ++d) acc += wr[d] * lr[d];
    s[idx] = acc * C_LIN + lb[c];
}

// ---------------------------------------------------------------------------
// 2) wq[o,i] = c_conv^2 * sum_{3x3} cw[o,i,.]^2   (for demodulation)
// ---------------------------------------------------------------------------
__global__ void wq_kernel(const float* __restrict__ cw,
                          float* __restrict__ q) {
    int idx = blockIdx.x * blockDim.x + threadIdx.x;   // o*512+i
    const float* p = cw + (size_t)idx * 9;
    float acc = 0.0f;
#pragma unroll
    for (int t = 0; t < 9; ++t) acc += p[t] * p[t];
    q[idx] = acc * C_CONV2;
}

// ---------------------------------------------------------------------------
// 3) sigma_inv[b,o] = rsqrt( sum_i s[b,i]^2 * q[o,i] + eps )
// ---------------------------------------------------------------------------
__global__ void sigma_kernel(const float* __restrict__ s,
                             const float* __restrict__ q,
                             float* __restrict__ sig) {
    int idx = blockIdx.x * blockDim.x + threadIdx.x;   // b*512+o
    int b = idx / COUT;
    int o = idx % COUT;
    const float* sr = s + (size_t)b * CIN;
    const float* qr = q + (size_t)o * CIN;
    float acc = 0.0f;
    for (int i = 0; i < CIN; ++i) {
        float si = sr[i];
        acc += si * si * qr[i];
    }
    sig[idx] = rsqrtf(acc + EPS);
}

// ---------------------------------------------------------------------------
// 4) modulated+demodulated weights, f16, layout wm[b][o][i*9+t] (K contiguous)
// ---------------------------------------------------------------------------
__global__ void modw_kernel(const float* __restrict__ cw,
                            const float* __restrict__ s,
                            const float* __restrict__ sig,
                            _Float16* __restrict__ wm) {
    size_t base = ((size_t)blockIdx.x * blockDim.x + threadIdx.x) * 8;
#pragma unroll
    for (int j = 0; j < 8; ++j) {
        size_t f = base + j;                    // < 8*512*4608
        int b = (int)(f / ((size_t)COUT * KTOT));
        int r = (int)(f % ((size_t)COUT * KTOT));
        int o = r / KTOT;
        int k = r % KTOT;
        int i = k / 9;
        int t = k % 9;
        float v = cw[((size_t)o * CIN + i) * 9 + t] * C_CONV
                * s[(size_t)b * CIN + i] * sig[(size_t)b * COUT + o];
        wm[f] = (_Float16)v;
    }
}

// ---------------------------------------------------------------------------
// 5) implicit-GEMM conv with WMMA f16 -> f32.
//    Block 256 threads = 8 waves; tile BM=128 x BN=256, BK=32.
//    Wave grid 2(M) x 4(N); each wave owns a 64x64 sub-tile = 4x4 WMMA tiles.
//    A-tile: double-buffered LDS, staged via global_load_async_to_lds_b128,
//            fenced with s_wait_asynccnt (ASYNCcnt, in-order completion).
//    B-tile: on-the-fly im2col; thread = one n-row, K index is a wave-uniform
//            literal so /9, %9, /3 land in SALU; only h/w adds+bounds in VALU.
//    Fragment loads: 2x ds_load_b128 per fragment per documented f16 layout
//      lanes 0-15 : k 0-7  (VGPR0-3), k 16-23 (VGPR4-7)
//      lanes 16-31: k 8-15 (VGPR0-3), k 24-31 (VGPR4-7)
// ---------------------------------------------------------------------------
#define BM 128
#define BN 256
#define BK 32
#define NIT (KTOT / BK)   // 144

union HV { v16h v; v8h h[2]; };

__global__ __launch_bounds__(256) void conv_wmma_kernel(
    const _Float16* __restrict__ wm,   // [B][COUT][KTOT]
    const float*    __restrict__ x,    // [B][CIN][64][64]
    float*          __restrict__ y)    // [B][COUT][64][64]
{
    __shared__ __align__(16) _Float16 As[2][BM * BK];  // 2 x 8 KB
    __shared__ __align__(16) _Float16 Bs[BN * BK];     // 16 KB, [n][k]

    const int b    = blockIdx.z;
    const int m0   = blockIdx.y * BM;
    const int n0   = blockIdx.x * BN;
    const int tid  = threadIdx.x;
    const int lane = tid & 31;
    const int wave = tid >> 5;
    const int wM   = wave & 1;      // x64 along M
    const int wN   = wave >> 1;     // x64 along N

    const v8f vzero = {0.f, 0.f, 0.f, 0.f, 0.f, 0.f, 0.f, 0.f};
    v8f acc[4][4];
#pragma unroll
    for (int mt = 0; mt < 4; ++mt)
#pragma unroll
        for (int nt = 0; nt < 4; ++nt) acc[mt][nt] = vzero;

    // ---- A staging: thread t -> 32 bytes: LDS halves [t*16, t*16+16) ----
    const int arow = tid >> 1;            // 0..127
    const int acol = (tid & 1) * 16;      // 0 or 16
    const _Float16* aglob = wm + ((size_t)b * COUT + m0 + arow) * KTOT + acol;

    // ---- B staging: thread t -> n-row t (all 32 k) ----
    const int gn = n0 + tid;
    const int oh = gn >> 6;
    const int ow = gn & 63;
    const float* xb = x + ((size_t)b * CIN << 12);

    // ---- fragment lane mapping (A and B tiles both [row][k]) ----
    const int frow = lane & 15;
    const int fk0  = (lane < 16) ? 0 : 8;

    // prologue: async-stage A tile for it=0 into buffer 0
    {
        unsigned lds0 = (unsigned)(uintptr_t)(&As[0][tid * 16]);
        unsigned long long ga = (unsigned long long)(uintptr_t)aglob;
        asm volatile("global_load_async_to_lds_b128 %0, %1, off"
                     :: "v"(lds0), "v"(ga) : "memory");
        asm volatile("global_load_async_to_lds_b128 %0, %1, off offset:16"
                     :: "v"(lds0), "v"(ga) : "memory");
    }

    int buf = 0;
    for (int it = 0; it < NIT; ++it) {
        const int k0 = it * BK;

        // ---- stage B tile: im2col, uniform k math in SALU ----
        {
            _Float16 tmp[BK];
#pragma unroll
            for (int kk = 0; kk < BK; ++kk) {
                int k  = k0 + kk;          // wave-uniform
                int ci = k / 9;            // SALU
                int t9 = k - ci * 9;       // SALU
                int kh = t9 / 3;           // SALU
                int kw = t9 - kh * 3;      // SALU
                int hs = oh + kh - 1;      // per-thread
                int ws = ow + kw - 1;
                float v = 0.0f;
                if ((unsigned)hs < 64u && (unsigned)ws < 64u)
                    v = xb[((size_t)ci << 12) + (hs << 6) + ws];
                tmp[kk] = (_Float16)v;
            }
#pragma unroll
            for (int seg = 0; seg < 4; ++seg)
                *(v8h*)(&Bs[tid * BK + seg * 8]) = *(v8h*)(&tmp[seg * 8]);
        }

        // ---- issue async A for next iteration into the other buffer ----
        if (it + 1 < NIT) {
            unsigned ldsn = (unsigned)(uintptr_t)(&As[buf ^ 1][tid * 16]);
            unsigned long long ga =
                (unsigned long long)(uintptr_t)(aglob + (size_t)(it + 1) * BK);
            asm volatile("global_load_async_to_lds_b128 %0, %1, off"
                         :: "v"(ldsn), "v"(ga) : "memory");
            asm volatile("global_load_async_to_lds_b128 %0, %1, off offset:16"
                         :: "v"(ldsn), "v"(ga) : "memory");
            // current tile's 2 async ops done; next tile's 2 still in flight
            asm volatile("s_wait_asynccnt 0x2" ::: "memory");
        } else {
            asm volatile("s_wait_asynccnt 0x0" ::: "memory");
        }
        __syncthreads();

        // ---- load fragments: 2x ds_load_b128 each ----
        const _Float16* Ab = &As[buf][0];
        v16h afrag[4], bfrag[4];
#pragma unroll
        for (int mt = 0; mt < 4; ++mt) {
            int row = wM * 64 + mt * 16 + frow;
            HV u;
            u.h[0] = *(const v8h*)(&Ab[row * BK + fk0]);
            u.h[1] = *(const v8h*)(&Ab[row * BK + fk0 + 16]);
            afrag[mt] = u.v;
        }
#pragma unroll
        for (int nt = 0; nt < 4; ++nt) {
            int row = wN * 64 + nt * 16 + frow;
            HV u;
            u.h[0] = *(const v8h*)(&Bs[row * BK + fk0]);
            u.h[1] = *(const v8h*)(&Bs[row * BK + fk0 + 16]);
            bfrag[nt] = u.v;
        }

        // ---- 16 WMMAs per wave per K-step ----
#pragma unroll
        for (int mt = 0; mt < 4; ++mt)
#pragma unroll
            for (int nt = 0; nt < 4; ++nt)
                acc[mt][nt] = __builtin_amdgcn_wmma_f32_16x16x32_f16(
                    false, afrag[mt], false, bfrag[nt],
                    (short)0, acc[mt][nt], false, false);

        __syncthreads();
        buf ^= 1;
    }

    // ---- epilogue: C/D layout lanes 0-15 M=r, lanes 16-31 M=8+r; N=lane%16
    const int moff = (lane < 16) ? 0 : 8;
    const int nL   = lane & 15;
#pragma unroll
    for (int mt = 0; mt < 4; ++mt)
#pragma unroll
        for (int nt = 0; nt < 4; ++nt) {
            int gm = m0 + wM * 64 + mt * 16 + moff;
            int gc = n0 + wN * 64 + nt * 16 + nL;
            float* yp = y + (((size_t)b * COUT + gm) << 12) + gc;
#pragma unroll
            for (int r = 0; r < 8; ++r)
                yp[(size_t)r << 12] = acc[mt][nt][r];
        }
}

// ---------------------------------------------------------------------------
// launch
// ---------------------------------------------------------------------------
extern "C" void kernel_launch(void* const* d_in, const int* in_sizes, int n_in,
                              void* d_out, int out_size, void* d_ws, size_t ws_size,
                              hipStream_t stream) {
    const float* x  = (const float*)d_in[0];   // [8,512,64,64]
    const float* w  = (const float*)d_in[1];   // [8,512]
    const float* cw = (const float*)d_in[2];   // [512,512,3,3]
    const float* lw = (const float*)d_in[3];   // [512,512]
    const float* lb = (const float*)d_in[4];   // [512]
    float* y = (float*)d_out;                  // [8,512,64,64]

    char* ws = (char*)d_ws;
    float*    s   = (float*)(ws);                              // 16 KB
    float*    sig = (float*)(ws + 16384);                      // 16 KB
    float*    q   = (float*)(ws + 32768);                      // 1 MB
    _Float16* wm  = (_Float16*)(ws + 32768 + 1048576);         // ~37.7 MB

    style_kernel<<<16,   256, 0, stream>>>(w, lw, lb, s);
    wq_kernel   <<<1024, 256, 0, stream>>>(cw, q);
    sigma_kernel<<<16,   256, 0, stream>>>(s, q, sig);
    modw_kernel <<<9216, 256, 0, stream>>>(cw, s, sig, wm);    // 18,874,368 elems

    dim3 grid(NTOT / BN, COUT / BM, BATCH);                    // (16, 4, 8)
    conv_wmma_kernel<<<grid, 256, 0, stream>>>(wm, x, y);
}
